// DeepFace_58033598104069
// MI455X (gfx1250) — compile-verified, matrix operand order checked by reading, not run on hardware
//
#include <hip/hip_runtime.h>

// ---------------------------------------------------------------------------
// DeepFace-style net on gfx1250 (wave32). All GEMM-shaped layers use
// v_wmma_f32_16x16x32_f16 with fp32 accumulation. Weight streams (read once,
// ~0.5 GB = the HBM roofline) go through per-wave double-buffered LDS via
// GLOBAL_LOAD_ASYNC_TO_LDS_B128 (ASYNCcnt). Intermediate activations are kept
// in f16 workspace: half the traffic and no per-consume cvt VALU.
// ---------------------------------------------------------------------------

#if __has_builtin(__builtin_amdgcn_global_load_async_to_lds_b128) && \
    __has_builtin(__builtin_amdgcn_s_wait_asynccnt)
#define HAVE_ASYNC_LDS 1
#else
#define HAVE_ASYNC_LDS 0
#endif

typedef __attribute__((ext_vector_type(16))) _Float16 v16h;
typedef __attribute__((ext_vector_type(8)))  _Float16 v8h;
typedef __attribute__((ext_vector_type(8)))  float    v8f;

__device__ __forceinline__ v8f wmma16(v16h a, v16h b, v8f c) {
  return __builtin_amdgcn_wmma_f32_16x16x32_f16(
      false, a, false, b, (short)0, c, false, false);
}

// K index at half-slot h (0..15) for lane-half hi:
// lanes 0-15:  {0..7, 16..23};  lanes 16-31: {8..15, 24..31}
__device__ __forceinline__ int frag_k(int h, int hi) {
  return h + ((h >= 8) ? 8 : 0) + (hi ? 8 : 0);
}

__device__ __forceinline__ v16h pack16(float4 f0, float4 f1, float4 f2, float4 f3) {
  v16h r;
  r[0]=(_Float16)f0.x;  r[1]=(_Float16)f0.y;  r[2]=(_Float16)f0.z;  r[3]=(_Float16)f0.w;
  r[4]=(_Float16)f1.x;  r[5]=(_Float16)f1.y;  r[6]=(_Float16)f1.z;  r[7]=(_Float16)f1.w;
  r[8]=(_Float16)f2.x;  r[9]=(_Float16)f2.y;  r[10]=(_Float16)f2.z; r[11]=(_Float16)f2.w;
  r[12]=(_Float16)f3.x; r[13]=(_Float16)f3.y; r[14]=(_Float16)f3.z; r[15]=(_Float16)f3.w;
  return r;
}

// f32 activation gather (conv1 input only)
__device__ __forceinline__ v16h gather16f(const float* __restrict__ base,
                                          int4 rA, int4 rB, int4 rC, int4 rD) {
  v16h r;
  r[0]=(_Float16)base[rA.x];  r[1]=(_Float16)base[rA.y];
  r[2]=(_Float16)base[rA.z];  r[3]=(_Float16)base[rA.w];
  r[4]=(_Float16)base[rB.x];  r[5]=(_Float16)base[rB.y];
  r[6]=(_Float16)base[rB.z];  r[7]=(_Float16)base[rB.w];
  r[8]=(_Float16)base[rC.x];  r[9]=(_Float16)base[rC.y];
  r[10]=(_Float16)base[rC.z]; r[11]=(_Float16)base[rC.w];
  r[12]=(_Float16)base[rD.x]; r[13]=(_Float16)base[rD.y];
  r[14]=(_Float16)base[rD.z]; r[15]=(_Float16)base[rD.w];
  return r;
}

// f16 activation gather: d16 loads pack straight into fragment regs, no cvt
__device__ __forceinline__ v16h gather16h(const _Float16* __restrict__ base,
                                          int4 rA, int4 rB, int4 rC, int4 rD) {
  v16h r;
  r[0]=base[rA.x];  r[1]=base[rA.y];  r[2]=base[rA.z];  r[3]=base[rA.w];
  r[4]=base[rB.x];  r[5]=base[rB.y];  r[6]=base[rB.z];  r[7]=base[rB.w];
  r[8]=base[rC.x];  r[9]=base[rC.y];  r[10]=base[rC.z]; r[11]=base[rC.w];
  r[12]=base[rD.x]; r[13]=base[rD.y]; r[14]=base[rD.z]; r[15]=base[rD.w];
  return r;
}

#if HAVE_ASYNC_LDS
// Builtin pointer parameter types per hipcc diagnostic:
//   src: __attribute__((vector_size(16))) int AS(1)*;  dst: same, AS(3)*
typedef int v4i __attribute__((vector_size(16)));
typedef __attribute__((address_space(1))) v4i* g4ptr;
typedef __attribute__((address_space(3))) v4i* l4ptr;

// Async-copy one 16-row x 32-float chunk (row stride = rowstride floats,
// starting at column kc*32) into a 512-float LDS buffer. 4x b128 per wave.
__device__ __forceinline__ void stage16x32(const float* __restrict__ src0,
                                           int rowstride, int kc,
                                           float* lbuf, int lane) {
#pragma unroll
  for (int i = 0; i < 4; ++i) {
    const int flat = i * 32 + lane;           // float4 index 0..127
    const int row = flat >> 3, c4 = flat & 7; // 8 float4 per row
    const float* src = src0 + row * rowstride + kc * 32 + c4 * 4;
    float* dst = lbuf + flat * 4;
    __builtin_amdgcn_global_load_async_to_lds_b128(
        (g4ptr)(void*)(const_cast<float*>(src)),
        (l4ptr)(void*)dst, 0, 0);
  }
}
#define ASYNC_WAIT(n) do { asm volatile("" ::: "memory"); \
  __builtin_amdgcn_s_wait_asynccnt(n); asm volatile("" ::: "memory"); } while (0)
#endif

// ---------------------------------------------------------------------------
// Pad conv1 weights (32,363) -> (32,384) zeros (aligned b128, guard-free loop)
// ---------------------------------------------------------------------------
__global__ __launch_bounds__(256) void k_padw1(const float* __restrict__ w,
                                               float* __restrict__ wpad) {
  const int t = blockIdx.x * blockDim.x + threadIdx.x;   // 12288
  const int n = t / 384, k = t % 384;
  wpad[t] = (k < 363) ? w[n * 363 + k] : 0.f;
}

// ---------------------------------------------------------------------------
// conv1: x(32,3,152,152) f32 -> y(32,32,142,142) f16, 11x11 VALID, ReLU
// One wave per 16-row M tile; both N tiles (32 oc) share one A fragment.
// ---------------------------------------------------------------------------
__global__ __launch_bounds__(256) void k_conv1(const float* __restrict__ x,
                                               const float* __restrict__ wpad,
                                               const float* __restrict__ bias,
                                               _Float16* __restrict__ y) {
  __shared__ __align__(16) int rel[384];
  for (int k = threadIdx.x; k < 384; k += 256) {
    const int kk = (k < 363) ? k : 0;
    const int c = kk / 121, rem = kk % 121;
    rel[k] = (c * 152 + rem / 11) * 152 + (rem % 11);
  }
  __syncthreads();

  const int wid  = blockIdx.x * (blockDim.x >> 5) + (threadIdx.x >> 5);
  const int lane = threadIdx.x & 31, ln = lane & 15, hi = lane >> 4;
  if (wid >= 40328) return;
  const int m  = wid * 16 + ln;
  const int b  = m / 20164, sp = m % 20164;
  const int oh = sp / 142,  ow = sp % 142;
  const float* xb = x + (b * 3 * 152 + oh) * 152 + ow;
  const int4*   rel4 = (const int4*)rel;
  const float4* w4   = (const float4*)wpad;
  const int n0 = ln, n1 = 16 + ln;

  v8f acc0 = {}, acc1 = {};
#pragma unroll 2
  for (int kc = 0; kc < 12; ++kc) {
    const int i0 = kc * 8 + hi * 2;
    const int4 rA = rel4[i0], rB = rel4[i0 + 1], rC = rel4[i0 + 4], rD = rel4[i0 + 5];
    const v16h a = gather16f(xb, rA, rB, rC, rD);
    const int wb0 = n0 * 96 + i0;
    const v16h bf0 = pack16(w4[wb0], w4[wb0 + 1], w4[wb0 + 4], w4[wb0 + 5]);
    const int wb1 = n1 * 96 + i0;
    const v16h bf1 = pack16(w4[wb1], w4[wb1 + 1], w4[wb1 + 4], w4[wb1 + 5]);
    acc0 = wmma16(a, bf0, acc0);
    acc1 = wmma16(a, bf1, acc1);
  }
  const float bb0 = bias[n0], bb1 = bias[n1];
#pragma unroll
  for (int r = 0; r < 8; ++r) {
    const int mr  = wid * 16 + r + hi * 8;
    const int br  = mr / 20164, spr = mr % 20164;
    const int ohr = spr / 142,  owr = spr % 142;
    const int o   = (br * 32 * 142 + ohr) * 142 + owr;
    y[o + n0 * 20164] = (_Float16)fmaxf(acc0[r] + bb0, 0.f);
    y[o + n1 * 20164] = (_Float16)fmaxf(acc1[r] + bb1, 0.f);
  }
}

// ---------------------------------------------------------------------------
// maxpool 3x3 stride 2 pad 1 (f16 in/out)
// ---------------------------------------------------------------------------
__global__ __launch_bounds__(256) void k_pool(const _Float16* __restrict__ y1,
                                              _Float16* __restrict__ p1) {
  const int t  = blockIdx.x * blockDim.x + threadIdx.x;   // < 5,161,984
  const int pw = t % 71;
  const int t1 = t / 71;
  const int ph = t1 % 71;
  const int bc = t1 / 71;
  float m = -1e30f;
#pragma unroll
  for (int dy = 0; dy < 3; ++dy) {
    const int r = ph * 2 - 1 + dy;
    if (r < 0 || r >= 142) continue;
#pragma unroll
    for (int dx = 0; dx < 3; ++dx) {
      const int cc = pw * 2 - 1 + dx;
      if (cc < 0 || cc >= 142) continue;
      m = fmaxf(m, (float)y1[(bc * 142 + r) * 142 + cc]);
    }
  }
  p1[t] = (_Float16)m;
}

// ---------------------------------------------------------------------------
// conv2: p1(32,32,71,71) f16 -> y(32,16,63,63) f16, 9x9 VALID, ReLU. K=2592.
// ---------------------------------------------------------------------------
__global__ __launch_bounds__(256) void k_conv2(const _Float16* __restrict__ p1,
                                               const float* __restrict__ w,
                                               const float* __restrict__ bias,
                                               _Float16* __restrict__ y) {
  __shared__ __align__(16) int rel[2592];
  for (int k = threadIdx.x; k < 2592; k += 256) {
    const int c = k / 81, rem = k % 81;
    rel[k] = (c * 71 + rem / 9) * 71 + (rem % 9);
  }
  __syncthreads();

  const int wid  = blockIdx.x * (blockDim.x >> 5) + (threadIdx.x >> 5);
  const int lane = threadIdx.x & 31, ln = lane & 15, hi = lane >> 4;
  if (wid >= 7938) return;
  const int m  = wid * 16 + ln;
  const int b  = m / 3969, sp = m % 3969;
  const int oh = sp / 63,  ow = sp % 63;
  const _Float16* pb = p1 + b * 32 * 5041 + oh * 71 + ow;
  const int4*   rel4 = (const int4*)rel;
  const float4* w4   = (const float4*)w;

  v8f acc = {};
#pragma unroll 2
  for (int kc = 0; kc < 81; ++kc) {
    const int i0 = kc * 8 + hi * 2;
    const int4 rA = rel4[i0], rB = rel4[i0 + 1], rC = rel4[i0 + 4], rD = rel4[i0 + 5];
    const v16h a = gather16h(pb, rA, rB, rC, rD);
    const int wb = ln * 648 + i0;
    const v16h bf = pack16(w4[wb], w4[wb + 1], w4[wb + 4], w4[wb + 5]);
    acc = wmma16(a, bf, acc);
  }
  const float bb = bias[ln];
#pragma unroll
  for (int r = 0; r < 8; ++r) {
    const int mr  = wid * 16 + r + hi * 8;
    const int br  = mr / 3969, spr = mr % 3969;
    const int ohr = spr / 63,  owr = spr % 63;
    y[((br * 16 + ln) * 63 + ohr) * 63 + owr] = (_Float16)fmaxf(acc[r] + bb, 0.f);
  }
}

// ---------------------------------------------------------------------------
// Local (unshared) conv (f16 activations): one wave per output position.
// M=32 (two tiles), N=16, K=16*KH*KW. Position weights async-staged via LDS.
// ---------------------------------------------------------------------------
template <int HIN, int WIN, int HOUT, int WOUT, int KH, int KW, int STRIDE>
__global__ __launch_bounds__(256) void k_lconv(const _Float16* __restrict__ in,
                                               const float* __restrict__ w,
                                               const float* __restrict__ bias,
                                               _Float16* __restrict__ out) {
  constexpr int K    = 16 * KH * KW;
  constexpr int KCF  = K / 32;
  constexpr int KPAD = ((K + 31) / 32) * 32;
  __shared__ __align__(16) int rel[KPAD];
#if HAVE_ASYNC_LDS
  __shared__ __align__(16) float wstage[8][2][512];   // per-wave double buffer
#endif
  for (int k = threadIdx.x; k < KPAD; k += 256) {
    const int kk = (k < K) ? k : 0;
    const int ch = kk / (KH * KW), rem = kk % (KH * KW);
    rel[k] = ch * (HIN * WIN) + (rem / KW) * WIN + (rem % KW);
  }
  __syncthreads();

  const int wid  = blockIdx.x * (blockDim.x >> 5) + (threadIdx.x >> 5);
  const int lane = threadIdx.x & 31, ln = lane & 15, hi = lane >> 4;
  const int wv   = threadIdx.x >> 5;
  if (wid >= HOUT * WOUT) return;
  const int p = wid / WOUT, q = wid % WOUT;
  const int posbase = (p * STRIDE) * WIN + q * STRIDE;
  const _Float16* in0 = in + ln * (16 * HIN * WIN) + posbase;
  const _Float16* in1 = in0 + 16 * (16 * HIN * WIN);
  const float* wp  = w + (size_t)wid * 16 * K;
  const int4* rel4 = (const int4*)rel;
#if !HAVE_ASYNC_LDS
  const float4* wp4 = (const float4*)wp;
#endif

  v8f c0 = {}, c1 = {};
#if HAVE_ASYNC_LDS
  stage16x32(wp, K, 0, &wstage[wv][0][0], lane);
#endif
  for (int kc = 0; kc < KCF; ++kc) {
    const int i0 = kc * 8 + hi * 2;
    const int4 rA = rel4[i0], rB = rel4[i0 + 1], rC = rel4[i0 + 4], rD = rel4[i0 + 5];
    const v16h a0 = gather16h(in0, rA, rB, rC, rD);
    const v16h a1 = gather16h(in1, rA, rB, rC, rD);
#if HAVE_ASYNC_LDS
    if (kc + 1 < KCF) {
      stage16x32(wp, K, kc + 1, &wstage[wv][(kc + 1) & 1][0], lane);
      ASYNC_WAIT(4);                        // chunk kc complete, kc+1 in flight
    } else {
      ASYNC_WAIT(0);
    }
    const float4* sb = (const float4*)&wstage[wv][kc & 1][0];
    const int wb = ln * 8 + hi * 2;
    const v16h bf = pack16(sb[wb], sb[wb + 1], sb[wb + 4], sb[wb + 5]);
#else
    __builtin_prefetch(wp + ln * K + (kc + 4) * 32, 0, 1);
    const int wb = ln * (K / 4) + i0;
    const v16h bf = pack16(wp4[wb], wp4[wb + 1], wp4[wb + 4], wp4[wb + 5]);
#endif
    c0 = wmma16(a0, bf, c0);
    c1 = wmma16(a1, bf, c1);
  }
  if (K % 32) {                             // guarded tail chunk (B only)
    const int i0 = KCF * 8 + hi * 2;
    const int4 rA = rel4[i0], rB = rel4[i0 + 1], rC = rel4[i0 + 4], rD = rel4[i0 + 5];
    const v16h a0 = gather16h(in0, rA, rB, rC, rD);
    const v16h a1 = gather16h(in1, rA, rB, rC, rD);
    v16h bf;
#pragma unroll
    for (int h = 0; h < 16; ++h) {
      const int k = KCF * 32 + frag_k(h, hi);
      bf[h] = (_Float16)((k < K) ? wp[ln * K + k] : 0.f);
    }
    c0 = wmma16(a0, bf, c0);
    c1 = wmma16(a1, bf, c1);
  }
  const float bb = bias[(ln * HOUT + p) * WOUT + q];
#pragma unroll
  for (int r = 0; r < 8; ++r) {
    const int b0 = r + hi * 8;
    out[((b0        * 16 + ln) * HOUT + p) * WOUT + q] = (_Float16)fmaxf(c0[r] + bb, 0.f);
    out[(((b0 + 16) * 16 + ln) * HOUT + p) * WOUT + q] = (_Float16)fmaxf(c1[r] + bb, 0.f);
  }
}

// ---------------------------------------------------------------------------
// fc1 partials: a(32,7056) f16 x W^T(f32) -> part(8,32,4096) f32; K split 8x.
// A fragments: 2x b128 half loads + shuffle (no cvt). W async-staged via LDS.
// ---------------------------------------------------------------------------
__global__ __launch_bounds__(256) void k_fc(const _Float16* __restrict__ a,
                                            const float* __restrict__ w,
                                            float* __restrict__ part) {
#if HAVE_ASYNC_LDS
  __shared__ __align__(16) float wstage[8][2][512];
#endif
  const int wid  = blockIdx.x * (blockDim.x >> 5) + (threadIdx.x >> 5);
  const int lane = threadIdx.x & 31, ln = lane & 15, hi = lane >> 4;
  const int wv   = threadIdx.x >> 5;
  if (wid >= 2048) return;
  const int tn  = wid & 255;
  const int seg = wid >> 8;
  const int n   = tn * 16 + ln;
  const int kc0 = seg * 28;
  const int kcV = (kc0 + 28 < 220) ? (kc0 + 28) : 220;
  const v8h*   a8 = (const v8h*)a;           // 8-half groups; 7056/8 = 882
  const float* wbase = w + tn * 16 * 7056;
#if !HAVE_ASYNC_LDS
  const float4* w4 = (const float4*)w;
#endif

  v8f c0 = {}, c1 = {};
#if HAVE_ASYNC_LDS
  stage16x32(wbase, 7056, kc0, &wstage[wv][kc0 & 1][0], lane);
#endif
  for (int kc = kc0; kc < kcV; ++kc) {
    const int g0 = ln * 882 + kc * 4 + hi;
    const v8h aL0 = a8[g0], aH0 = a8[g0 + 2];
    const v16h af0 = __builtin_shufflevector(aL0, aH0, 0,1,2,3,4,5,6,7,
                                             8,9,10,11,12,13,14,15);
    const int g1 = (ln + 16) * 882 + kc * 4 + hi;
    const v8h aL1 = a8[g1], aH1 = a8[g1 + 2];
    const v16h af1 = __builtin_shufflevector(aL1, aH1, 0,1,2,3,4,5,6,7,
                                             8,9,10,11,12,13,14,15);
#if HAVE_ASYNC_LDS
    if (kc + 1 < kcV) {
      stage16x32(wbase, 7056, kc + 1, &wstage[wv][(kc + 1) & 1][0], lane);
      ASYNC_WAIT(4);
    } else {
      ASYNC_WAIT(0);
    }
    const float4* sb = (const float4*)&wstage[wv][kc & 1][0];
    const int wb = ln * 8 + hi * 2;
    const v16h bf = pack16(sb[wb], sb[wb + 1], sb[wb + 4], sb[wb + 5]);
#else
    __builtin_prefetch(w + n * 7056 + (kc + 4) * 32, 0, 1);
    const int wb = n * 1764 + kc * 8 + hi * 2;
    const v16h bf = pack16(w4[wb], w4[wb + 1], w4[wb + 4], w4[wb + 5]);
#endif
    c0 = wmma16(af0, bf, c0);
    c1 = wmma16(af1, bf, c1);
  }
  if (seg == 7) {                           // guarded tail chunk kc=220
    v16h af0, af1, bf;
#pragma unroll
    for (int h = 0; h < 16; ++h) {
      const int k = 220 * 32 + frag_k(h, hi);
      const bool ok = (k < 7056);
      af0[h] = ok ? a[ln * 7056 + k] : (_Float16)0.f;
      af1[h] = ok ? a[(ln + 16) * 7056 + k] : (_Float16)0.f;
      bf[h]  = (_Float16)(ok ? w[n * 7056 + k] : 0.f);
    }
    c0 = wmma16(af0, bf, c0);
    c1 = wmma16(af1, bf, c1);
  }
#pragma unroll
  for (int r = 0; r < 8; ++r) {
    const int m0 = r + hi * 8;
    part[(seg * 32 + m0)      * 4096 + n] = c0[r];
    part[(seg * 32 + m0 + 16) * 4096 + n] = c1[r];
  }
}

// Deterministic reduction of K-segments + bias
__global__ __launch_bounds__(256) void k_reduce(const float* __restrict__ part,
                                                const float* __restrict__ bias,
                                                float* __restrict__ out) {
  const int t = blockIdx.x * blockDim.x + threadIdx.x;  // 131072
  const int n = t & 4095;
  const int m = t >> 12;
  float s = bias[n];
#pragma unroll
  for (int seg = 0; seg < 8; ++seg) s += part[(seg * 32 + m) * 4096 + n];
  out[t] = s;
}

// ---------------------------------------------------------------------------
extern "C" void kernel_launch(void* const* d_in, const int* in_sizes, int n_in,
                              void* d_out, int out_size, void* d_ws, size_t ws_size,
                              hipStream_t stream) {
  (void)in_sizes; (void)n_in; (void)out_size; (void)ws_size;
  const float* x   = (const float*)d_in[0];
  const float* w1  = (const float*)d_in[1];
  const float* b1  = (const float*)d_in[2];
  const float* w2  = (const float*)d_in[3];
  const float* b2  = (const float*)d_in[4];
  const float* lw1 = (const float*)d_in[5];
  const float* lb1 = (const float*)d_in[6];
  const float* lw2 = (const float*)d_in[7];
  const float* lb2 = (const float*)d_in[8];
  const float* lw3 = (const float*)d_in[9];
  const float* lb3 = (const float*)d_in[10];
  const float* fw  = (const float*)d_in[11];
  const float* fb  = (const float*)d_in[12];
  float* out = (float*)d_out;

  // Workspace (f16 activations). Region A (20,647,936 halves) holds y1 then
  // y2..y5; region B (5,161,984 halves) holds p1 then fc partials (f32).
  _Float16* wsh = (_Float16*)d_ws;
  _Float16* y1  = wsh;                        // (32,32,142,142) f16
  _Float16* p1  = wsh + 20647936;             // (32,32,71,71)  f16
  _Float16* y2  = wsh;                        // reuse region A
  _Float16* y3  = y2 + 2032128;
  _Float16* y4  = y3 + 1548800;
  _Float16* y5  = y4 + 320000;
  float*    part = (float*)p1;                // 4 MB, reuse region B
  float*    w1p  = (float*)(p1 + 5161984);    // (32,384) f32

  k_padw1<<<48, 256, 0, stream>>>(w1, w1p);
  k_conv1<<<5041, 256, 0, stream>>>(x, w1p, b1, y1);
  k_pool<<<20164, 256, 0, stream>>>(y1, p1);
  k_conv2<<<993, 256, 0, stream>>>(p1, w2, b2, y2);
  k_lconv<63, 63, 55, 55, 9, 9, 1><<<379, 256, 0, stream>>>(y2, lw1, lb1, y3);
  k_lconv<55, 55, 25, 25, 7, 7, 2><<<79,  256, 0, stream>>>(y3, lw2, lb2, y4);
  k_lconv<25, 25, 21, 21, 5, 5, 1><<<56,  256, 0, stream>>>(y4, lw3, lb3, y5);
  k_fc<<<256, 256, 0, stream>>>(y5, fw, part);
  k_reduce<<<512, 256, 0, stream>>>(part, fb, out);
}
